// GIN_16312285790930
// MI455X (gfx1250) — matrix operand order repaired and checked by reading, not compiled, not run on hardware
//
#include <hip/hip_runtime.h>
#include <stdint.h>

#define N_NODES  200000
#define N_EDGES  400000
#define CH       128
#define N_GRAPHS 8192
#define BN_EPS   1e-5f

typedef __bf16 bf16_t;
typedef bf16_t v16bf __attribute__((ext_vector_type(16)));
typedef float  v8f   __attribute__((ext_vector_type(8)));

union FragAB { v16bf v; unsigned int u[8]; };

__device__ __forceinline__ unsigned short f32_bf16(float f) {
    unsigned int u = __float_as_uint(f);
    u += 0x7FFFu + ((u >> 16) & 1u);     // round-to-nearest-even
    return (unsigned short)(u >> 16);
}

// ---------------- prep kernels ----------------

// transpose + convert one 128x128 f32 weight (row-major [k][n]) to bf16 [n][k]
__global__ void wt_prep(const float* __restrict__ w, unsigned short* __restrict__ wt) {
    int i = blockIdx.x * blockDim.x + threadIdx.x;
    if (i < CH * CH) {
        int k = i >> 7, n = i & 127;
        wt[n * CH + k] = f32_bf16(w[k * CH + n]);
    }
}

// fold BN (eval) into scale/shift per channel
__global__ void bn_prep(const float* __restrict__ g, const float* __restrict__ b,
                        const float* __restrict__ m, const float* __restrict__ v,
                        float* __restrict__ s, float* __restrict__ t) {
    int i = threadIdx.x;
    float sv = g[i] * rsqrtf(v[i] + BN_EPS);
    s[i] = sv;
    t[i] = b[i] - m[i] * sv;
}

// ---------------- data-movement kernels ----------------

__global__ void copy_f32v4(const float4* __restrict__ src, float4* __restrict__ dst, int n4) {
    int i = blockIdx.x * blockDim.x + threadIdx.x;
    if (i < n4) dst[i] = src[i];
}

__global__ void zero_f32(float* __restrict__ p, int n) {
    int i = blockIdx.x * blockDim.x + threadIdx.x;
    if (i < n) p[i] = 0.f;
}

// agg[dst] += feat[src], one thread per (edge, channel)
__global__ void edge_scatter(const float* __restrict__ feat, float* __restrict__ agg,
                             const int* __restrict__ esrc, const int* __restrict__ edst) {
    int tid = blockIdx.x * blockDim.x + threadIdx.x;
    int e = tid >> 7;
    int c = tid & 127;
    if (e < N_EDGES) {
        atomicAdd(&agg[(size_t)edst[e] * CH + c], feat[(size_t)esrc[e] * CH + c]);
    }
}

// P[batch[node]] += H[node], one thread per (node, channel)
__global__ void pool_scatter(const float* __restrict__ H, const int* __restrict__ batch,
                             float* __restrict__ P) {
    int tid = blockIdx.x * blockDim.x + threadIdx.x;
    int node = tid >> 7;
    int c = tid & 127;
    if (node < N_NODES) {
        atomicAdd(&P[(size_t)batch[node] * CH + c], H[(size_t)node * CH + c]);
    }
}

// ---------------- WMMA MLP kernel ----------------
// A-fragment (16x32 bf16) from LDS, per ISA 16-bit A layout:
// lanes 0-15: M=lane, VGPR v holds K = 16*(v/4) + (v%4)*2 {+1}
// lanes 16-31: same M, K offset +8
__device__ __forceinline__ void frag_a_lds(FragAB& a, const unsigned short* lbuf,
                                           int lane, int kt) {
    int m = lane & 15, g = lane >> 4;
    const unsigned short* row = lbuf + m * CH;
#pragma unroll
    for (int v = 0; v < 8; ++v) {
        int kb = kt * 32 + (v >> 2) * 16 + g * 8 + (v & 3) * 2;
        a.u[v] = *(const unsigned int*)&row[kb];
    }
}

// B-fragment (32x16 bf16) from bf16 weight stored [n][k]:
// lanes 0-15 hold K=0..7,16..23 of the slab (2 per VGPR), lanes 16-31 hold K=8..15,24..31
__device__ __forceinline__ void frag_b_glb(FragAB& b, const unsigned short* __restrict__ wT,
                                           int lane, int nt, int kt) {
    int n = nt * 16 + (lane & 15), g = lane >> 4;
    const unsigned int* p = (const unsigned int*)&wT[n * CH + kt * 32 + g * 8];
#pragma unroll
    for (int v = 0; v < 8; ++v) b.u[v] = p[(v & 3) + (v >> 2) * 8];
}

__global__ void __launch_bounds__(128)
gin_mlp(const float* __restrict__ X,              // (x + aggr), N_NODES x 128 f32
        const unsigned short* __restrict__ w1T,   // bf16 [n][k]
        const float* __restrict__ b1,
        const float* __restrict__ bnS, const float* __restrict__ bnT,
        const unsigned short* __restrict__ w2T,   // bf16 [n][k]
        const float* __restrict__ b2,
        float* __restrict__ Y) {
    __shared__ unsigned short sA[4][16 * CH];
    __shared__ unsigned short sH[4][16 * CH];

    const int wave = threadIdx.x >> 5;
    const int lane = threadIdx.x & 31;
    const int tile = blockIdx.x * 4 + wave;   // N_NODES/16 tiles, exact
    const int row0 = tile * 16;

    unsigned short* lA = sA[wave];
    unsigned short* lH = sH[wave];

    // stage 16x128 rows: f32 -> bf16 LDS (float4 per lane, 32 lanes cover a row)
    {
        const float4* Xv = (const float4*)(X + (size_t)row0 * CH);
#pragma unroll 4
        for (int m = 0; m < 16; ++m) {
            float4 f = Xv[m * 32 + lane];
            int base = m * CH + lane * 4;
            lA[base + 0] = f32_bf16(f.x);
            lA[base + 1] = f32_bf16(f.y);
            lA[base + 2] = f32_bf16(f.z);
            lA[base + 3] = f32_bf16(f.w);
        }
    }
    __syncthreads();

    // ---- GEMM 1: H1 = relu(BN(A @ W1 + b1)), bf16 result staged to LDS ----
    for (int nt = 0; nt < 8; ++nt) {
        v8f acc = {0.f, 0.f, 0.f, 0.f, 0.f, 0.f, 0.f, 0.f};
#pragma unroll
        for (int kt = 0; kt < 4; ++kt) {
            FragAB a, b;
            frag_a_lds(a, lA, lane, kt);
            frag_b_glb(b, w1T, lane, nt, kt);
            acc = __builtin_amdgcn_wmma_f32_16x16x32_bf16(
                false, a.v, false, b.v, (short)0, acc, false, false);
        }
        int n = nt * 16 + (lane & 15);
        float s = bnS[n], t = bnT[n], bb = b1[n];
#pragma unroll
        for (int r = 0; r < 8; ++r) {
            int m = (lane >> 4) * 8 + r;       // C/D layout: lane=N, VGPR r -> M
            float h = fmaxf((acc[r] + bb) * s + t, 0.f);
            lH[m * CH + n] = f32_bf16(h);
        }
    }
    __syncthreads();

    // ---- GEMM 2: Y = relu(H1 @ W2 + b2), f32 to global ----
    for (int nt = 0; nt < 8; ++nt) {
        v8f acc = {0.f, 0.f, 0.f, 0.f, 0.f, 0.f, 0.f, 0.f};
#pragma unroll
        for (int kt = 0; kt < 4; ++kt) {
            FragAB a, b;
            frag_a_lds(a, lH, lane, kt);
            frag_b_glb(b, w2T, lane, nt, kt);
            acc = __builtin_amdgcn_wmma_f32_16x16x32_bf16(
                false, a.v, false, b.v, (short)0, acc, false, false);
        }
        int n = nt * 16 + (lane & 15);
        float bb = b2[n];
#pragma unroll
        for (int r = 0; r < 8; ++r) {
            int m = (lane >> 4) * 8 + r;
            Y[(size_t)(row0 + m) * CH + n] = fmaxf(acc[r] + bb, 0.f);
        }
    }
}

// ---------------- final projection: out[g] = dot(P[g], lin_w) + lin_b ----------------
__global__ void final_lin(const float* __restrict__ P, const float* __restrict__ lw,
                          const float* __restrict__ lb, float* __restrict__ out) {
    int wave = threadIdx.x >> 5, lane = threadIdx.x & 31;
    int gidx = blockIdx.x * 8 + wave;
    if (gidx >= N_GRAPHS) return;
    const float* row = P + (size_t)gidx * CH;
    float s = 0.f;
#pragma unroll
    for (int k = lane; k < CH; k += 32) s += row[k] * lw[k];
#pragma unroll
    for (int off = 16; off; off >>= 1) s += __shfl_down(s, off, 32);
    if (lane == 0) out[gidx] = s + lb[0];
}

// ---------------- host launch ----------------
extern "C" void kernel_launch(void* const* d_in, const int* in_sizes, int n_in,
                              void* d_out, int out_size, void* d_ws, size_t ws_size,
                              hipStream_t stream) {
    const float* x     = (const float*)d_in[0];
    const int*   ei    = (const int*)d_in[1];
    const int*   batch = (const int*)d_in[2];
    const float* lin_w = (const float*)d_in[3];
    const float* lin_b = (const float*)d_in[4];
    const float* L[3][8];   // w1,b1,gamma,beta,mean,var,w2,b2
    for (int l = 0; l < 3; ++l)
        for (int j = 0; j < 8; ++j)
            L[l][j] = (const float*)d_in[5 + l * 8 + j];

    char* ws = (char*)d_ws;
    size_t off = 0;
    float* G = (float*)(ws + off); off += (size_t)N_NODES * CH * 4;   // aggregate buffer
    float* H = (float*)(ws + off); off += (size_t)N_NODES * CH * 4;   // layer output
    float* P = (float*)(ws + off); off += (size_t)N_GRAPHS * CH * 4;  // pooled
    unsigned short* WT = (unsigned short*)(ws + off); off += (size_t)6 * CH * CH * 2;
    float* BNS = (float*)(ws + off); off += 3 * CH * 4;
    float* BNT = (float*)(ws + off); off += 3 * CH * 4;
    (void)ws_size; (void)in_sizes; (void)n_in; (void)out_size;

    const int* esrc = ei;
    const int* edst = ei + N_EDGES;

    // fold weights (bf16 transposed) and BN params
    for (int l = 0; l < 3; ++l) {
        wt_prep<<<(CH * CH + 255) / 256, 256, 0, stream>>>(L[l][0], WT + (size_t)(2 * l) * CH * CH);
        wt_prep<<<(CH * CH + 255) / 256, 256, 0, stream>>>(L[l][6], WT + (size_t)(2 * l + 1) * CH * CH);
        bn_prep<<<1, CH, 0, stream>>>(L[l][2], L[l][3], L[l][4], L[l][5], BNS + l * CH, BNT + l * CH);
    }

    const int copyBlocks = (N_NODES * CH / 4 + 255) / 256;
    const int scatBlocks = (N_EDGES * CH + 255) / 256;
    const int mlpBlocks  = N_NODES / 16 / 4;     // 3125 blocks x 4 waves

    const float* feat = x;
    for (int l = 0; l < 3; ++l) {
        // G = feat; G[dst] += feat[src]  =>  G = feat + aggr
        copy_f32v4<<<copyBlocks, 256, 0, stream>>>((const float4*)feat, (float4*)G, N_NODES * CH / 4);
        edge_scatter<<<scatBlocks, 256, 0, stream>>>(feat, G, esrc, edst);
        gin_mlp<<<mlpBlocks, 128, 0, stream>>>(G,
                                               WT + (size_t)(2 * l) * CH * CH, L[l][1],
                                               BNS + l * CH, BNT + l * CH,
                                               WT + (size_t)(2 * l + 1) * CH * CH, L[l][7],
                                               H);
        feat = H;
    }

    zero_f32<<<(N_GRAPHS * CH + 255) / 256, 256, 0, stream>>>(P, N_GRAPHS * CH);
    pool_scatter<<<(N_NODES * CH + 255) / 256, 256, 0, stream>>>(H, batch, P);
    final_lin<<<(N_GRAPHS + 7) / 8, 256, 0, stream>>>(P, lin_w, lin_b, (float*)d_out);
}
